// APPNP_56556129354474
// MI455X (gfx1250) — matrix operand order repaired and checked by reading, not compiled, not run on hardware
//
#include <hip/hip_runtime.h>
#include <hip/hip_bf16.h>

typedef __attribute__((ext_vector_type(16))) _Float16 v16h;
typedef __attribute__((ext_vector_type(8)))  float    v8f;

#define N_NODES 100000
#define N_EDGES 1600000
#define K_STEPS 10
#define ALPHA_F 0.1f
#define D_IN    512
#define D_HID   256
#define D_OUT   64

// ---------------------------------------------------------------------------
// Small utility kernels
// ---------------------------------------------------------------------------
__global__ void zero_f32(float* __restrict__ p, int n) {
    int i = blockIdx.x * blockDim.x + threadIdx.x;
    if (i < n) p[i] = 0.f;
}

// W [K x Nn] f32 row-major  ->  WT [Nn x K] f16 row-major
__global__ void transpose_to_f16(const float* __restrict__ W, _Float16* __restrict__ WT,
                                 int K, int Nn) {
    int i = blockIdx.x * blockDim.x + threadIdx.x;
    if (i < K * Nn) {
        int k = i / Nn;
        int n = i - k * Nn;
        WT[(size_t)n * K + k] = (_Float16)W[i];
    }
}

__global__ void degree_kernel(const int* __restrict__ src, const int* __restrict__ dst,
                              float* __restrict__ outd, float* __restrict__ ind, int E) {
    int i = blockIdx.x * blockDim.x + threadIdx.x;
    if (i < E) {
        atomicAdd(&outd[src[i]], 1.f);
        atomicAdd(&ind[dst[i]], 1.f);
    }
}

__global__ void norm_kernel(float* __restrict__ outd, float* __restrict__ ind, int n) {
    int i = blockIdx.x * blockDim.x + threadIdx.x;
    if (i < n) {
        outd[i] = 1.f / sqrtf(fmaxf(outd[i], 1.f));
        ind[i]  = 1.f / sqrtf(fmaxf(ind[i],  1.f));
    }
}

// ---------------------------------------------------------------------------
// WMMA GEMM:  C[M x Nn] = act(A[M x K] * WT[Nn x K]^T + bias)
//  - A: row-major, f32 or f16 (converted to f16 in registers)
//  - WT: weights pre-transposed to [Nn x K] f16 so each B fragment
//    (one column of B per lane) is a contiguous 32-byte row read.
//  - one wave computes a 16x64 tile: 1 A fragment reused over 4 WMMAs.
//  - block = 128 threads = 4 waves stacked along M.
// A fragment (16x32 f16) per-lane layout (contiguous k, same order as B):
//   lanes 0-15 : row = lane,    halfs 0..15 = K(k+0..15)
//   lanes 16-31: row = lane-16, halfs 0..15 = K(k+16..31)
// B fragment (32x16 f16):
//   lanes 0-15 : col = lane,    halfs = K(k+0..15)
//   lanes 16-31: col = lane-16, halfs = K(k+16..31)
// C/D (16x16 f32): lane&15 = col, VGPR j = row j (+8 for lanes 16-31)
// ---------------------------------------------------------------------------
template <typename AT, typename OT, bool RELU>
__global__ __launch_bounds__(128)
void wmma_gemm_kernel(const AT* __restrict__ A, const _Float16* __restrict__ WT,
                      const float* __restrict__ bias, OT* __restrict__ C,
                      int M, int K, int Nn) {
    const int lane = threadIdx.x & 31;
    const int warp = threadIdx.x >> 5;
    const int row  = (blockIdx.x * 4 + warp) * 16;
    if (row >= M) return;                      // wave-uniform: EXEC stays all-1s
    const int ncol = blockIdx.y * 64;

    const int r    = row + (lane & 15);
    const int akb  = (lane >> 4) << 4;         // +16 halfs for upper half-wave
    const int bkb  = (lane >> 4) << 4;         // +16 halfs for upper half-wave
    const int bn   = ncol + (lane & 15);

    v8f acc[4] = {};

    for (int k = 0; k < K; k += 32) {
        // --- A fragment: two contiguous 8-half windows ---
        const AT* ap = A + (size_t)r * K + (k + akb);
        v16h a;
#pragma unroll
        for (int i = 0; i < 8; ++i) a[i]     = (_Float16)ap[i];
#pragma unroll
        for (int i = 0; i < 8; ++i) a[i + 8] = (_Float16)ap[8 + i];

        // --- 4 B fragments, 4 WMMAs (A reused) ---
#pragma unroll
        for (int t = 0; t < 4; ++t) {
            const _Float16* bp = WT + (size_t)(bn + 16 * t) * K + (k + bkb);
            v16h b;
#pragma unroll
            for (int i = 0; i < 16; ++i) b[i] = bp[i];
            acc[t] = __builtin_amdgcn_wmma_f32_16x16x32_f16(
                false, a, false, b, (short)0, acc[t], false, false);
        }
    }

    // --- epilogue: bias (+ReLU) and scattered stores per C layout ---
    const int r0 = row + ((lane >> 4) << 3);
#pragma unroll
    for (int t = 0; t < 4; ++t) {
        const int c = ncol + 16 * t + (lane & 15);
        const float bv = bias[c];
#pragma unroll
        for (int j = 0; j < 8; ++j) {
            float v = acc[t][j] + bv;
            if (RELU) v = fmaxf(v, 0.f);
            C[(size_t)(r0 + j) * Nn + c] = (OT)v;
        }
    }
}

// ---------------------------------------------------------------------------
// Propagation: warp-per-edge scatter (lane owns 2 of the 64 features)
// ---------------------------------------------------------------------------
__global__ void scatter_kernel(const float* __restrict__ h, const int* __restrict__ src,
                               const int* __restrict__ dst, const float* __restrict__ src_norm,
                               float* __restrict__ agg, int E) {
    int gid  = blockIdx.x * blockDim.x + threadIdx.x;
    int e    = gid >> 5;
    int lane = gid & 31;
    if (e >= E) return;
    int s = src[e];
    int d = dst[e];
    float nrm = src_norm[s];
    const float2 v = *(const float2*)(h + (size_t)s * 64 + lane * 2);
    float* ap = agg + (size_t)d * 64 + lane * 2;
    atomicAdd(ap + 0, v.x * nrm);
    atomicAdd(ap + 1, v.y * nrm);
}

// h_out = (1-alpha) * agg * dst_norm + alpha * h0 ; agg re-zeroed for next step
__global__ void combine_kernel(float* __restrict__ agg, const float* __restrict__ dst_norm,
                               const float* __restrict__ h0, float* __restrict__ hout,
                               int total) {
    int i = blockIdx.x * blockDim.x + threadIdx.x;
    if (i < total) {
        int node = i >> 6;
        float a  = agg[i];
        agg[i]   = 0.f;
        hout[i]  = (1.f - ALPHA_F) * a * dst_norm[node] + ALPHA_F * h0[i];
    }
}

// ---------------------------------------------------------------------------
extern "C" void kernel_launch(void* const* d_in, const int* in_sizes, int n_in,
                              void* d_out, int out_size, void* d_ws, size_t ws_size,
                              hipStream_t stream) {
    const float* feats = (const float*)d_in[0];
    const int*   src   = (const int*)d_in[1];
    const int*   dst   = (const int*)d_in[2];
    const float* W1    = (const float*)d_in[3];
    const float* b1    = (const float*)d_in[4];
    const float* W2    = (const float*)d_in[5];
    const float* b2    = (const float*)d_in[6];
    float* out = (float*)d_out;

    // --- carve workspace (256B aligned slices) ---
    char* ws = (char*)d_ws;
    size_t off = 0;
    auto carve = [&](size_t bytes) -> char* {
        char* p = ws + off;
        off = (off + bytes + 255) & ~(size_t)255;
        return p;
    };
    float*    agg  = (float*)   carve((size_t)N_NODES * D_OUT * 4);
    float*    outd = (float*)   carve((size_t)N_NODES * 4);          // -> src_norm
    float*    ind  = (float*)   carve((size_t)N_NODES * 4);          // -> dst_norm
    float*    h0   = (float*)   carve((size_t)N_NODES * D_OUT * 4);  // MLP output
    float*    hA   = (float*)   carve((size_t)N_NODES * D_OUT * 4);
    float*    hB   = (float*)   carve((size_t)N_NODES * D_OUT * 4);
    _Float16* W1T  = (_Float16*)carve((size_t)D_HID * D_IN * 2);
    _Float16* W2T  = (_Float16*)carve((size_t)D_OUT * D_HID * 2);
    _Float16* h1   = (_Float16*)carve((size_t)N_NODES * D_HID * 2);

    const int TPB = 256;

    // --- init / weight prep ---
    zero_f32<<<(N_NODES * D_OUT + TPB - 1) / TPB, TPB, 0, stream>>>(agg, N_NODES * D_OUT);
    zero_f32<<<(N_NODES + TPB - 1) / TPB, TPB, 0, stream>>>(outd, N_NODES);
    zero_f32<<<(N_NODES + TPB - 1) / TPB, TPB, 0, stream>>>(ind, N_NODES);
    transpose_to_f16<<<(D_IN * D_HID + TPB - 1) / TPB, TPB, 0, stream>>>(W1, W1T, D_IN, D_HID);
    transpose_to_f16<<<(D_HID * D_OUT + TPB - 1) / TPB, TPB, 0, stream>>>(W2, W2T, D_HID, D_OUT);

    // --- degrees + norms ---
    degree_kernel<<<(N_EDGES + TPB - 1) / TPB, TPB, 0, stream>>>(src, dst, outd, ind, N_EDGES);
    norm_kernel<<<(N_NODES + TPB - 1) / TPB, TPB, 0, stream>>>(outd, ind, N_NODES);

    // --- MLP via WMMA ---
    const int mstrips = (N_NODES / 16 + 3) / 4;  // waves of 16 rows, 4 waves/block
    {
        dim3 g1(mstrips, D_HID / 64);
        wmma_gemm_kernel<float, _Float16, true><<<g1, 128, 0, stream>>>(
            feats, W1T, b1, h1, N_NODES, D_IN, D_HID);
    }
    {
        dim3 g2(mstrips, D_OUT / 64);
        wmma_gemm_kernel<_Float16, float, false><<<g2, 128, 0, stream>>>(
            h1, W2T, b2, h0, N_NODES, D_HID, D_OUT);
    }

    // --- K-step APPNP propagation (L2-resident: h/agg = 25.6 MB each) ---
    const int sc_blocks = ((N_EDGES * 32) + TPB - 1) / TPB;
    const int cb_blocks = (N_NODES * D_OUT + TPB - 1) / TPB;
    const float* h_in = h0;
    for (int k = 0; k < K_STEPS; ++k) {
        float* h_out = (k == K_STEPS - 1) ? out : ((k & 1) ? hB : hA);
        scatter_kernel<<<sc_blocks, TPB, 0, stream>>>(h_in, src, dst, outd, agg, N_EDGES);
        combine_kernel<<<cb_blocks, TPB, 0, stream>>>(agg, ind, h0, h_out, N_NODES * D_OUT);
        h_in = h_out;
    }
}
